// MultiHeadLatentAttention_2138893714150
// MI455X (gfx1250) — matrix operand order repaired and checked by reading, not compiled, not run on hardware
//
#include <hip/hip_runtime.h>
#include <cmath>
#include <cstdint>
#include <cstddef>
#include <type_traits>

// ---------------------------------------------------------------------------
// MLA forward for MI455X (gfx1250, wave32, WMMA + TDM async tensor loads).
//   B=2, S=2048, E=2048, H=16, DH=128, LOW=512, R=64, Dqk=DH+R=192
// ---------------------------------------------------------------------------

typedef __attribute__((ext_vector_type(16))) __bf16        bf16x16;
typedef __attribute__((ext_vector_type(8)))  __bf16        bf16x8;
typedef __attribute__((ext_vector_type(4)))  __bf16        bf16x4;
typedef __attribute__((ext_vector_type(8)))  float         f32x8;
typedef __attribute__((ext_vector_type(4)))  float         f32x4;
typedef __attribute__((ext_vector_type(4)))  unsigned int  u32x4;
typedef __attribute__((ext_vector_type(8)))  int           i32x8;
typedef __attribute__((ext_vector_type(4)))  int           i32x4;

#define MLA_B  2
#define MLA_S  2048
#define MLA_E  2048
#define MLA_H  16
#define MLA_DH 128
#define MLA_R  64
#define MLA_DQK 192

static __device__ __forceinline__ bf16x16 bfcat(bf16x8 lo, bf16x8 hi) {
    return __builtin_shufflevector(lo, hi, 0,1,2,3,4,5,6,7,8,9,10,11,12,13,14,15);
}

static __device__ __forceinline__ f32x8 wmma_bf16(bf16x16 a, bf16x16 b, f32x8 c) {
    return __builtin_amdgcn_wmma_f32_16x16x32_bf16(
        /*neg_a=*/false, a, /*neg_b=*/false, b,
        /*c_mod=*/(short)0, c, /*reuse_a=*/false, /*reuse_b=*/false);
}

// ---------------------------------------------------------------------------
// TDM: async 2D bf16 tile load Global -> LDS via Tensor Data Mover.
// Descriptor bitfields per CDNA5 ISA.
// This toolchain exposes the 6-arg builtin:
//   (u32x4 group0, i32x8 group1, i32x4 group2, i32x4 group3, i32x8, i32 cpol)
// Groups 2/3 (+trailing i32x8) are zero: 2D tensor, no gather/iterate.
// Issue from ONE wave; pair with s_wait_tensorcnt + workgroup barrier.
//   tile_w/tensor_w/row_stride in elements (data_size = 2 bytes).
//   pad: pad_interval code (2^code * 2 DWORDs stored per pad), pad_amount code
//        (code+1 DWORDs of LDS padding inserted) -- load only.
// ---------------------------------------------------------------------------
static __device__ __forceinline__ void tdm_load_2d_bf16(
    uint32_t lds_off, const void* gptr,
    uint32_t tile_w, uint32_t tile_h,
    uint32_t tensor_w, uint32_t tensor_h, uint32_t row_stride,
    bool pad_en, uint32_t pad_interval, uint32_t pad_amount) {
    uint64_t ga = (uint64_t)(uintptr_t)gptr;
    u32x4 g0;
    g0[0] = 1u;                                   // count=1 valid user descriptor
    g0[1] = lds_off;                              // lds_addr (bytes)
    g0[2] = (uint32_t)ga;                         // global_addr[31:0]
    g0[3] = (uint32_t)(ga >> 32) | (2u << 30);    // global_addr[56:32] | type=2
    i32x8 g1;
    uint32_t w0 = (1u << 16);                     // data_size=1 (2B), wg_mask=0
    if (pad_en) w0 |= (1u << 20) | (pad_interval << 22) | (pad_amount << 25);
    g1[0] = (int)w0;
    g1[1] = (int)((tensor_w & 0xFFFFu) << 16);                              // dim0[15:0]
    g1[2] = (int)(((tensor_w >> 16) & 0xFFFFu) | ((tensor_h & 0xFFFFu) << 16)); // dim0[31:16]|dim1[15:0]
    g1[3] = (int)(((tensor_h >> 16) & 0xFFFFu) | ((tile_w & 0xFFFFu) << 16));   // dim1[31:16]|tile0
    g1[4] = (int)(tile_h & 0xFFFFu);                                        // tile1 (tile2=0)
    g1[5] = (int)row_stride;                                                // dim0_stride[31:0]
    g1[6] = 0;                                                              // stride[47:32], d1str lo
    g1[7] = 0;
    i32x4 z4 = (i32x4){0, 0, 0, 0};
    i32x8 z8 = (i32x8){0, 0, 0, 0, 0, 0, 0, 0};
    __builtin_amdgcn_tensor_load_to_lds(g0, g1, z4, z4, z8, 0);
}

// ---------------------------------------------------------------------------
// Tiled GEMM: C[M,N] = A[M,K] (row-major) * B[K,N] (row-major fp32 weights)
// Block tile 128x128, K-step 64, 8 waves (4 along M x 2 along N),
// each wave owns 32x64 of C via 2x4 WMMA tiles; 16 WMMAs per barrier pair.
// bf16 A tiles come in via the TDM (async, LDS-padded); fp32 A and the
// fp32->bf16 transposed B tiles use vector loads + LDS scatter.
// ---------------------------------------------------------------------------
template <typename TA, typename TOut>
__global__ __launch_bounds__(256)
void gemm_wmma_kernel(const TA* __restrict__ A, const float* __restrict__ Bw,
                      TOut* __restrict__ C, int M, int N, int K) {
    __shared__ __bf16 As[128][72];    // [m][k] pad 8 (144B rows, 16B aligned)
    __shared__ __bf16 Bst[128][72];   // transposed: [n][k]

    const int tid  = threadIdx.x;
    const int lane = tid & 31;
    const int wid  = tid >> 5;
    const int l15  = lane & 15;
    const int hl   = lane >> 4;
    const int wm   = (wid & 3) * 32;
    const int wn   = (wid >> 2) * 64;
    const int m0   = blockIdx.y * 128;
    const int n0   = blockIdx.x * 128;

    f32x8 acc[2][4];
#pragma unroll
    for (int mt = 0; mt < 2; ++mt)
#pragma unroll
        for (int nt = 0; nt < 4; ++nt) acc[mt][nt] = (f32x8){};

    const uint32_t as_off = (uint32_t)(uintptr_t)&As[0][0];

    for (int k0 = 0; k0 < K; k0 += 64) {
        __syncthreads();
        // ---- A tile (128 x 64) ----
        if constexpr (std::is_same_v<TA, __bf16>) {
            if (wid == 0) {
                // async TDM load; pad 4 DWORDs after every 32 DWORDs (row=128B)
                tdm_load_2d_bf16(as_off, A + (size_t)m0 * K + k0,
                                 /*tile_w=*/64, /*tile_h=*/128,
                                 /*tensor_w=*/(uint32_t)K, /*tensor_h=*/(uint32_t)M,
                                 /*row_stride=*/(uint32_t)K,
                                 /*pad=*/true, /*interval=*/4, /*amount=*/3);
            }
        } else {
#pragma unroll
            for (int i = 0; i < 8; ++i) {
                int idx = i * 1024 + tid * 4;
                int row = idx >> 6, col = idx & 63;
                f32x4 a4 = *(const f32x4*)(A + (size_t)(m0 + row) * K + k0 + col);
                bf16x4 b4;
#pragma unroll
                for (int j = 0; j < 4; ++j) b4[j] = (__bf16)a4[j];
                *(bf16x4*)&As[row][col] = b4;
            }
        }
        // ---- B tile (64 x 128) fp32 -> bf16, transposed into LDS ----
#pragma unroll
        for (int i = 0; i < 8; ++i) {
            int idx = i * 1024 + tid * 4;
            int krow = idx >> 7, col = idx & 127;
            int gcol = n0 + col;
            f32x4 b4 = (f32x4){};
            if (gcol < N) b4 = *(const f32x4*)(Bw + (size_t)(k0 + krow) * N + gcol);
#pragma unroll
            for (int j = 0; j < 4; ++j) Bst[col + j][krow] = (__bf16)b4[j];
        }
        // prefetch next B k-tile into L2 while WMMAs run
        if (k0 + 64 < K) {
            int pcol = n0 + (tid & 127);
            if (pcol < N)
                __builtin_prefetch(Bw + (size_t)(k0 + 64 + (tid >> 7) * 32) * N + pcol, 0, 1);
        }
        if constexpr (std::is_same_v<TA, __bf16>) {
            if (wid == 0) __builtin_amdgcn_s_wait_tensorcnt(0);
        }
        __syncthreads();

        // ---- fragments + WMMA: 2 K-chunks of 32 ----
#pragma unroll
        for (int c = 0; c < 2; ++c) {
            bf16x16 afrag[2], bfrag[4];
            const int akb = c * 32 + hl * 8;
#pragma unroll
            for (int mt = 0; mt < 2; ++mt) {
                int r = wm + mt * 16 + l15;
                bf16x8 lo = *(const bf16x8*)&As[r][akb];
                bf16x8 hi = *(const bf16x8*)&As[r][akb + 16];
                afrag[mt] = bfcat(lo, hi);
            }
            const int bkb = c * 32 + hl * 8;
#pragma unroll
            for (int nt = 0; nt < 4; ++nt) {
                int cc = wn + nt * 16 + l15;
                bf16x8 lo = *(const bf16x8*)&Bst[cc][bkb];
                bf16x8 hi = *(const bf16x8*)&Bst[cc][bkb + 16];
                bfrag[nt] = bfcat(lo, hi);
            }
#pragma unroll
            for (int mt = 0; mt < 2; ++mt)
#pragma unroll
                for (int nt = 0; nt < 4; ++nt)
                    acc[mt][nt] = wmma_bf16(afrag[mt], bfrag[nt], acc[mt][nt]);
        }
    }

    // ---- store C (row M = hl*8 + r, col N = l15 per ISA C-layout) ----
#pragma unroll
    for (int mt = 0; mt < 2; ++mt)
#pragma unroll
        for (int nt = 0; nt < 4; ++nt)
#pragma unroll
            for (int r = 0; r < 8; ++r) {
                int row = m0 + wm + mt * 16 + hl * 8 + r;
                int col = n0 + wn + nt * 16 + l15;
                if (col < N) C[(size_t)row * N + col] = (TOut)acc[mt][nt][r];
            }
}

// ---------------------------------------------------------------------------
// Assemble q / k: dst[b,h,s,0:128] = Csrc[(b*S+s), h*128 + d]
//                 dst[b,h,s,128:192] = RoPE(Rsrc row)
// ---------------------------------------------------------------------------
__global__ __launch_bounds__(256)
void assemble_qk_kernel(const __bf16* __restrict__ Csrc,
                        const __bf16* __restrict__ Rsrc,
                        __bf16* __restrict__ dst,
                        int rope_row_stride, int rope_head_stride) {
    size_t idx = (size_t)blockIdx.x * blockDim.x + threadIdx.x;
    const size_t total = (size_t)MLA_B * MLA_H * MLA_S * MLA_DQK;
    if (idx >= total) return;
    int d = (int)(idx % MLA_DQK);
    size_t t = idx / MLA_DQK;
    int s = (int)(t % MLA_S); t /= MLA_S;
    int h = (int)(t % MLA_H);
    int b = (int)(t / MLA_H);
    size_t row = (size_t)b * MLA_S + s;
    __bf16 outv;
    if (d < MLA_DH) {
        outv = Csrc[row * MLA_E + h * MLA_DH + d];
    } else {
        int j  = d - MLA_DH;          // 0..63
        int jj = (j < 32) ? j : j - 32;
        const __bf16* rp = Rsrc + row * rope_row_stride + h * rope_head_stride;
        float x1 = (float)rp[2 * jj];
        float x2 = (float)rp[2 * jj + 1];
        float th = (float)s * __expf(-(float)jj * 0.28782313662425575f); // 10000^(-jj/32)
        float sn, cs;
        __sincosf(th, &sn, &cs);
        outv = (__bf16)((j < 32) ? (x1 * cs - x2 * sn) : (x1 * sn + x2 * cs));
    }
    dst[idx] = outv;
}

__global__ __launch_bounds__(256)
void assemble_v_kernel(const __bf16* __restrict__ vC, __bf16* __restrict__ dst) {
    size_t idx = (size_t)blockIdx.x * blockDim.x + threadIdx.x;
    const size_t total = (size_t)MLA_B * MLA_H * MLA_S * MLA_DH;
    if (idx >= total) return;
    int d = (int)(idx % MLA_DH);
    size_t t = idx / MLA_DH;
    int s = (int)(t % MLA_S); t /= MLA_S;
    int h = (int)(t % MLA_H);
    int b = (int)(t / MLA_H);
    dst[idx] = vC[((size_t)b * MLA_S + s) * MLA_E + h * MLA_DH + d];
}

// ---------------------------------------------------------------------------
// Causal flash attention. One block = 128 query rows of one (b,h);
// 8 waves x 16 rows. 32-key tiles: K tile via TDM (async, TENSORcnt),
// V tile transposed manually. Online softmax fp32; ctx bf16 [B*S, H*128].
// ---------------------------------------------------------------------------
__global__ __launch_bounds__(256)
void mla_flash_attn_kernel(const __bf16* __restrict__ Q,
                           const __bf16* __restrict__ Kc,
                           const __bf16* __restrict__ V,
                           __bf16* __restrict__ Octx, float scale) {
    __shared__ __bf16 Kt[32][MLA_DQK]; // TDM dest: packed rows [krow][dqk]
    __shared__ __bf16 VtT[128][40];    // value tile transposed [dv][krow]
    __shared__ __bf16 Pls[8][16][40];  // per-wave P staging (16x32)

    const int qblk = blockIdx.x;
    const int h    = blockIdx.y;
    const int b    = blockIdx.z;
    const int tid  = threadIdx.x;
    const int lane = tid & 31;
    const int wid  = tid >> 5;
    const int l15  = lane & 15;
    const int hl   = lane >> 4;
    const int qbase = qblk * 128 + wid * 16;

    const __bf16* Qbh = Q  + ((size_t)(b * MLA_H + h)) * MLA_S * MLA_DQK;
    const __bf16* Kbh = Kc + ((size_t)(b * MLA_H + h)) * MLA_S * MLA_DQK;
    const __bf16* Vbh = V  + ((size_t)(b * MLA_H + h)) * MLA_S * MLA_DH;
    const uint32_t kt_off = (uint32_t)(uintptr_t)&Kt[0][0];

    // ---- Q fragments in registers: 6 chunks of K=32 ----
    bf16x16 qf[6];
    {
        const __bf16* qrow = Qbh + (size_t)(qbase + l15) * MLA_DQK;
#pragma unroll
        for (int c = 0; c < 6; ++c) {
            int kb = c * 32 + hl * 8;
            bf16x8 lo = *(const bf16x8*)(qrow + kb);
            bf16x8 hi = *(const bf16x8*)(qrow + kb + 16);
            qf[c] = bfcat(lo, hi);
        }
    }

    f32x8 oacc[8];
#pragma unroll
    for (int nt = 0; nt < 8; ++nt) oacc[nt] = (f32x8){};
    float rm[8], rl[8];
#pragma unroll
    for (int r = 0; r < 8; ++r) { rm[r] = -__builtin_inff(); rl[r] = 0.0f; }

    const int nkt = (qblk + 1) * 4;
    for (int kt = 0; kt < nkt; ++kt) {
        const int kb0 = kt * 32;
        __syncthreads();
        // ---- K tile 32x192 via TDM (one wave issues the DMA) ----
        if (wid == 0) {
            tdm_load_2d_bf16(kt_off, Kbh + (size_t)kb0 * MLA_DQK,
                             /*tile_w=*/MLA_DQK, /*tile_h=*/32,
                             /*tensor_w=*/MLA_DQK, /*tensor_h=*/MLA_S,
                             /*row_stride=*/MLA_DQK,
                             /*pad=*/false, 0, 0);
        }
        // ---- V tile 32x128 transposed (manual) ----
#pragma unroll
        for (int i = 0; i < 2; ++i) {
            int idx = tid + i * 256;
            int row = idx >> 4, c8 = (idx & 15) * 8;
            bf16x8 vv = *(const bf16x8*)(Vbh + (size_t)(kb0 + row) * MLA_DH + c8);
#pragma unroll
            for (int j = 0; j < 8; ++j) VtT[c8 + j][row] = vv[j];
        }
        // prefetch next tiles while this one is consumed
        if (kt + 1 < nkt) {
            __builtin_prefetch(Kbh + (size_t)(kb0 + 32 + (tid >> 5) * 4) * MLA_DQK + (tid & 31) * 6, 0, 1);
            __builtin_prefetch(Vbh + (size_t)(kb0 + 32 + (tid >> 4)) * MLA_DH + (tid & 15) * 8, 0, 1);
        }
        if (wid == 0) __builtin_amdgcn_s_wait_tensorcnt(0);
        __syncthreads();

        // ---- S = Q * K^T : 2 N-tiles of 16 keys, 6 chained WMMAs each ----
        f32x8 sacc[2];
        sacc[0] = (f32x8){}; sacc[1] = (f32x8){};
#pragma unroll
        for (int nt = 0; nt < 2; ++nt) {
            int krow = nt * 16 + l15;
#pragma unroll
            for (int c = 0; c < 6; ++c) {
                int kk = c * 32 + hl * 8;
                bf16x8 lo = *(const bf16x8*)&Kt[krow][kk];
                bf16x8 hi = *(const bf16x8*)&Kt[krow][kk + 16];
                sacc[nt] = wmma_bf16(qf[c], bfcat(lo, hi), sacc[nt]);
            }
        }

        // ---- online softmax over this 32-key tile ----
#pragma unroll
        for (int r = 0; r < 8; ++r) {
            int qrow = qbase + hl * 8 + r;
#pragma unroll
            for (int nt = 0; nt < 2; ++nt) {
                int kcol = kb0 + nt * 16 + l15;
                float sv = sacc[nt][r] * scale;
                if (kcol > qrow) sv = -__builtin_inff();
                sacc[nt][r] = sv;
            }
            float mx = fmaxf(sacc[0][r], sacc[1][r]);
#pragma unroll
            for (int off = 8; off >= 1; off >>= 1)
                mx = fmaxf(mx, __shfl_xor(mx, off, 32));
            float m_new = fmaxf(rm[r], mx);
            float a = __expf(rm[r] - m_new);
            float ssum = 0.0f;
#pragma unroll
            for (int nt = 0; nt < 2; ++nt) {
                float p = __expf(sacc[nt][r] - m_new);
                sacc[nt][r] = p;
                ssum += p;
            }
#pragma unroll
            for (int off = 8; off >= 1; off >>= 1)
                ssum += __shfl_xor(ssum, off, 32);
            rl[r] = rl[r] * a + ssum;
            rm[r] = m_new;
#pragma unroll
            for (int nt = 0; nt < 8; ++nt) oacc[nt][r] *= a;
        }

        // ---- P: C-layout -> A-layout via wave-private LDS ----
#pragma unroll
        for (int r = 0; r < 8; ++r) {
            Pls[wid][hl * 8 + r][l15]      = (__bf16)sacc[0][r];
            Pls[wid][hl * 8 + r][16 + l15] = (__bf16)sacc[1][r];
        }
        bf16x16 pf;
        {
            int kb = hl * 8;
            bf16x8 lo = *(const bf16x8*)&Pls[wid][l15][kb];
            bf16x8 hi = *(const bf16x8*)&Pls[wid][l15][kb + 16];
            pf = bfcat(lo, hi);
        }

        // ---- O += P * V ----
#pragma unroll
        for (int nt = 0; nt < 8; ++nt) {
            int cc = nt * 16 + l15;
            int kb = hl * 8;
            bf16x8 lo = *(const bf16x8*)&VtT[cc][kb];
            bf16x8 hi = *(const bf16x8*)&VtT[cc][kb + 16];
            oacc[nt] = wmma_bf16(pf, bfcat(lo, hi), oacc[nt]);
        }
    }

    // ---- normalize + store ctx as [B*S, H*128] bf16 ----
#pragma unroll
    for (int r = 0; r < 8; ++r) {
        int qrow = qbase + hl * 8 + r;
        float inv = 1.0f / rl[r];
#pragma unroll
        for (int nt = 0; nt < 8; ++nt) {
            int col = nt * 16 + l15;
            Octx[((size_t)(b * MLA_S + qrow)) * MLA_E + h * MLA_DH + col] =
                (__bf16)(oacc[nt][r] * inv);
        }
    }
}

// ---------------------------------------------------------------------------
// Host launch
// ---------------------------------------------------------------------------
extern "C" void kernel_launch(void* const* d_in, const int* in_sizes, int n_in,
                              void* d_out, int out_size, void* d_ws, size_t ws_size,
                              hipStream_t stream) {
    const float* x     = (const float*)d_in[0];
    const float* W_DQ  = (const float*)d_in[1];
    const float* W_UQ  = (const float*)d_in[2];
    const float* W_QR  = (const float*)d_in[3];
    const float* W_DKV = (const float*)d_in[4];
    const float* W_UK  = (const float*)d_in[5];
    const float* W_KR  = (const float*)d_in[6];
    const float* W_UV  = (const float*)d_in[7];
    const float* W_O   = (const float*)d_in[8];
    float* out = (float*)d_out;

    const int M = MLA_B * MLA_S;  // 4096 rows

    char* w = (char*)d_ws;
    auto carve = [&](size_t bytes) {
        char* p = w;
        w += (bytes + 255) & ~(size_t)255;
        return p;
    };
    __bf16* cQ  = (__bf16*)carve((size_t)M * MLA_E * 2);
    __bf16* qC  = (__bf16*)carve((size_t)M * MLA_E * 2);
    __bf16* qR  = (__bf16*)carve((size_t)M * (MLA_R * MLA_H) * 2);
    __bf16* cKV = (__bf16*)carve((size_t)M * 512 * 2);
    __bf16* kC  = (__bf16*)carve((size_t)M * MLA_E * 2);
    __bf16* kR  = (__bf16*)carve((size_t)M * MLA_R * 2);
    __bf16* vC  = (__bf16*)carve((size_t)M * MLA_E * 2);
    __bf16* q   = (__bf16*)carve((size_t)MLA_B * MLA_H * MLA_S * MLA_DQK * 2);
    __bf16* k   = (__bf16*)carve((size_t)MLA_B * MLA_H * MLA_S * MLA_DQK * 2);
    __bf16* v   = (__bf16*)carve((size_t)MLA_B * MLA_H * MLA_S * MLA_DH * 2);
    __bf16* ctx = (__bf16*)carve((size_t)M * MLA_E * 2);

    const dim3 blk(256);
    auto grid2 = [](int n, int m) { return dim3((n + 127) / 128, (m + 127) / 128); };

    // 1) c_Q = x @ W_DQ
    gemm_wmma_kernel<float, __bf16><<<grid2(MLA_E, M), blk, 0, stream>>>(x, W_DQ, cQ, M, MLA_E, MLA_E);
    // 2) q_C = c_Q @ W_UQ
    gemm_wmma_kernel<__bf16, __bf16><<<grid2(MLA_E, M), blk, 0, stream>>>(cQ, W_UQ, qC, M, MLA_E, MLA_E);
    // 3) q_R(raw) = c_Q @ W_QR
    gemm_wmma_kernel<__bf16, __bf16><<<grid2(MLA_R * MLA_H, M), blk, 0, stream>>>(cQ, W_QR, qR, M, MLA_R * MLA_H, MLA_E);
    // 4) c_KV = x @ W_DKV
    gemm_wmma_kernel<float, __bf16><<<grid2(512, M), blk, 0, stream>>>(x, W_DKV, cKV, M, 512, MLA_E);
    // 5) k_C = c_KV @ W_UK
    gemm_wmma_kernel<__bf16, __bf16><<<grid2(MLA_E, M), blk, 0, stream>>>(cKV, W_UK, kC, M, MLA_E, 512);
    // 6) k_R(raw) = x @ W_KR
    gemm_wmma_kernel<float, __bf16><<<grid2(MLA_R, M), blk, 0, stream>>>(x, W_KR, kR, M, MLA_R, MLA_E);
    // 7) v_C = c_KV @ W_UV
    gemm_wmma_kernel<__bf16, __bf16><<<grid2(MLA_E, M), blk, 0, stream>>>(cKV, W_UV, vC, M, MLA_E, 512);

    // 8) assemble q / k (content + RoPE) and v into [B,H,S,*] bf16
    {
        size_t tqk = (size_t)MLA_B * MLA_H * MLA_S * MLA_DQK;
        size_t tv  = (size_t)MLA_B * MLA_H * MLA_S * MLA_DH;
        assemble_qk_kernel<<<dim3((unsigned)((tqk + 255) / 256)), blk, 0, stream>>>(
            qC, qR, q, MLA_R * MLA_H, MLA_R);
        assemble_qk_kernel<<<dim3((unsigned)((tqk + 255) / 256)), blk, 0, stream>>>(
            kC, kR, k, MLA_R, 0);
        assemble_v_kernel<<<dim3((unsigned)((tv + 255) / 256)), blk, 0, stream>>>(vC, v);
    }

    // 9) causal flash attention -> ctx [B*S, H*128]
    {
        float scale = 1.0f / sqrtf((float)(MLA_DH + MLA_R));
        mla_flash_attn_kernel<<<dim3(MLA_S / 128, MLA_H, MLA_B), blk, 0, stream>>>(
            q, k, v, ctx, scale);
    }

    // 10) out = ctx @ W_O -> fp32 d_out
    gemm_wmma_kernel<__bf16, float><<<grid2(MLA_E, M), blk, 0, stream>>>(ctx, W_O, out, M, MLA_E, MLA_E);
}
